// EncDec_5789615915124
// MI455X (gfx1250) — matrix operand order repaired and checked by reading, not compiled, not run on hardware
//
#include <hip/hip_runtime.h>

typedef __attribute__((ext_vector_type(16))) _Float16 v16h;
typedef __attribute__((ext_vector_type(8)))  _Float16 v8h;
typedef __attribute__((ext_vector_type(8)))  float    v8f;

#define H        512
#define G4       2048            // 4*H gates
#define TSTEPS   64
#define BATCH    1024
#define BROWS    16              // batch rows per block
#define NBLK     (BATCH / BROWS) // 64 persistent blocks
#define NTHREADS 256
#define NWAVES   (NTHREADS / 32)
#define K0       544             // h0(512) + x(4) + zero pad(28)
#define K1       1056            // h0(512) + xpad(32) + h1(512)
#define KCH0     (K0 / 32)       // 17 k-chunks
#define KCH1     (K1 / 32)       // 33 k-chunks
#define XOFF     512
#define H1OFF    544
#define NTILES   (G4 / 16)       // 128 column tiles of 16
#define FRAGH    512             // halves per packed B fragment (32 lanes x 16)

struct SharedBufs {
  _Float16 h[BROWS][K1];      // [h0 | x,pad | h1] per row, f16 (A operand)
  float    gates[BROWS][G4];  // f32 gate pre-activations (also proj scratch)
  float    c[2][BROWS][H];    // cell state, both layers
};

__device__ __forceinline__ float sigmoidf_(float v) {
  return 1.0f / (1.0f + __expf(-v));
}

// gates[BROWS][G4] = s.h[:, 0:kdim] @ W^T
// W is packed fragment-major: fragment f = tile*kchunks + kc occupies
// halves [f*512, f*512+512); lane l's 16 halves at f*512 + l*16 hold
// column n = tile*16 + (l&15), K = kc*32 + (l>>4)*16 + i  (contiguous k order, same as A).
__device__ __forceinline__ void wmma_matmul(
    const _Float16* __restrict__ W, int kchunks,
    const _Float16* __restrict__ hbase, float (*__restrict__ gates)[G4]) {
  const int lane  = threadIdx.x & 31;
  const int wave  = threadIdx.x >> 5;
  const int m     = lane & 15;
  const int ka    = (lane >> 4) * 16;  // A: halves 0..15 -> K=ka..ka+15
  const int rbase = (lane >> 4) * 8;   // C: vgpr r -> row rbase+r
  const _Float16* arow = hbase + m * K1;
  for (int tile = wave; tile < NTILES; tile += NWAVES) {
    const _Float16* wfrag =
        W + (size_t)(tile * kchunks) * FRAGH + lane * 16;
    v8f acc = {};
    for (int kc = 0; kc < kchunks; ++kc) {
      const int k0 = kc << 5;
      union { v16h v; v8h h[2]; } a, b;
      a.h[0] = *(const v8h*)(arow + k0 + ka);        // ds_load_b128
      a.h[1] = *(const v8h*)(arow + k0 + ka + 8);    // ds_load_b128
      b.v    = *(const v16h*)(wfrag + (size_t)kc * FRAGH); // coalesced 1KB/wave
      acc = __builtin_amdgcn_wmma_f32_16x16x32_f16(
          false, a.v, false, b.v, (short)0, acc, false, false);
    }
    const int ncol = tile * 16 + m;
#pragma unroll
    for (int r = 0; r < 8; ++r)
      gates[rbase + r][ncol] = acc[r];
  }
}

__device__ __forceinline__ void lstm_elem(
    SharedBufs& s, const float* __restrict__ bias, int layer, int hoff) {
  for (int idx = threadIdx.x; idx < BROWS * H; idx += NTHREADS) {
    const int r = idx >> 9;
    const int j = idx & (H - 1);
    float gi = s.gates[r][j]         + bias[j];
    float gf = s.gates[r][H + j]     + bias[H + j];
    float gg = s.gates[r][2 * H + j] + bias[2 * H + j];
    float go = s.gates[r][3 * H + j] + bias[3 * H + j];
    gi = sigmoidf_(gi); gf = sigmoidf_(gf); go = sigmoidf_(go);
    gg = tanhf(gg);
    float cv = gf * s.c[layer][r][j] + gi * gg;
    s.c[layer][r][j] = cv;
    s.h[r][hoff + j] = (_Float16)(go * tanhf(cv));
  }
}

// ---- weight conversion: f32 -> f16, packed WMMA-fragment-major ----
// dst element at (f, l, i): f = frag = tile*kchunks + kc,
//   n = tile*16 + (l&15),  k = kc*32 + (l>>4)*16 + i
__global__ __launch_bounds__(256)
void cvt_w0cat(const float* __restrict__ whh0, const float* __restrict__ wih0,
               _Float16* __restrict__ dst) {
  int idx = blockIdx.x * 256 + threadIdx.x;
  if (idx >= G4 * K0) return;
  int f = idx >> 9, rem = idx & 511;
  int l = rem >> 4, i = rem & 15;
  int tile = f / KCH0, kc = f - tile * KCH0;
  int n = tile * 16 + (l & 15);
  int k = kc * 32 + (l >> 4) * 16 + i;
  float v = 0.0f;
  if (k < H)          v = whh0[n * H + k];
  else if (k < H + 4) v = wih0[n * 4 + (k - H)];
  dst[idx] = (_Float16)v;
}

__global__ __launch_bounds__(256)
void cvt_w1cat(const float* __restrict__ wih1, const float* __restrict__ whh1,
               _Float16* __restrict__ dst) {
  int idx = blockIdx.x * 256 + threadIdx.x;
  if (idx >= G4 * K1) return;
  int f = idx >> 9, rem = idx & 511;
  int l = rem >> 4, i = rem & 15;
  int tile = f / KCH1, kc = f - tile * KCH1;
  int n = tile * 16 + (l & 15);
  int k = kc * 32 + (l >> 4) * 16 + i;
  float v = 0.0f;
  if (k < H)            v = wih1[n * H + k];
  else if (k >= H1OFF)  v = whh1[n * H + (k - H1OFF)];
  dst[idx] = (_Float16)v;
}

// ---- persistent encoder/decoder kernel: one block = 16 batch rows ----
__global__ __launch_bounds__(NTHREADS, 1)
void lstm_encdec_kernel(
    const float* __restrict__ x,
    const float* __restrict__ enc_b0, const float* __restrict__ enc_b1,
    const float* __restrict__ dec_b0, const float* __restrict__ dec_b1,
    const _Float16* __restrict__ encW0, const _Float16* __restrict__ encW1,
    const _Float16* __restrict__ decW0, const _Float16* __restrict__ decW1,
    const float* __restrict__ lin_w, const float* __restrict__ lin_b,
    float* __restrict__ out) {
  __shared__ SharedBufs s;
  const int tid   = threadIdx.x;
  const int brow0 = blockIdx.x * BROWS;

  for (int i = tid; i < BROWS * K1; i += NTHREADS)
    (&s.h[0][0])[i] = (_Float16)0.0f;
  for (int i = tid; i < 2 * BROWS * H; i += NTHREADS)
    (&s.c[0][0][0])[i] = 0.0f;
  __syncthreads();

  // ---------------- encoder ----------------
  for (int t = 0; t < TSTEPS; ++t) {
    if (tid < BROWS * 4) {
      int r = tid >> 2, d = tid & 3;
      s.h[r][XOFF + d] =
          (_Float16)x[((size_t)(brow0 + r) * TSTEPS + t) * 4 + d];
    }
    __syncthreads();
    wmma_matmul(encW0, KCH0, &s.h[0][0], s.gates);
    __syncthreads();
    lstm_elem(s, enc_b0, 0, 0);
    __syncthreads();
    wmma_matmul(encW1, KCH1, &s.h[0][0], s.gates);
    __syncthreads();
    lstm_elem(s, enc_b1, 1, H1OFF);
    __syncthreads();
  }

  // ---------------- decoder (autoregressive, seeded with x[:,0,:]) ----
  if (tid < BROWS * 4) {
    int r = tid >> 2, d = tid & 3;
    s.h[r][XOFF + d] = (_Float16)x[(size_t)(brow0 + r) * TSTEPS * 4 + d];
  }
  __syncthreads();
  for (int t = 0; t < TSTEPS; ++t) {
    wmma_matmul(decW0, KCH0, &s.h[0][0], s.gates);
    __syncthreads();
    lstm_elem(s, dec_b0, 0, 0);
    __syncthreads();
    wmma_matmul(decW1, KCH1, &s.h[0][0], s.gates);
    __syncthreads();
    lstm_elem(s, dec_b1, 1, H1OFF);
    __syncthreads();
    // out = h1 @ lin_w^T + lin_b : 64 dots of length 512, 4 threads each
    {
      int p = tid >> 2, q = tid & 3;          // pair p in [0,64), quarter q
      int r = p >> 2, d = p & 3;
      float part = 0.0f;
      int j0 = q * (H / 4);
      for (int j = j0; j < j0 + H / 4; ++j)
        part += (float)s.h[r][H1OFF + j] * lin_w[d * H + j];
      (&s.gates[0][0])[tid] = part;           // gates buffer is dead here
    }
    __syncthreads();
    if (tid < BROWS * 4) {
      int r = tid >> 2, d = tid & 3;
      const float* part = &s.gates[0][0] + tid * 4;
      float acc = lin_b[d] + part[0] + part[1] + part[2] + part[3];
      out[((size_t)(brow0 + r) * TSTEPS + t) * 4 + d] = acc;
      s.h[r][XOFF + d] = (_Float16)acc;       // next step's input
    }
    __syncthreads();
  }
}

extern "C" void kernel_launch(void* const* d_in, const int* in_sizes, int n_in,
                              void* d_out, int out_size, void* d_ws, size_t ws_size,
                              hipStream_t stream) {
  const float* x       = (const float*)d_in[0];
  const float* encWih0 = (const float*)d_in[1];
  const float* encWhh0 = (const float*)d_in[2];
  const float* enc_b0  = (const float*)d_in[3];
  const float* encWih1 = (const float*)d_in[4];
  const float* encWhh1 = (const float*)d_in[5];
  const float* enc_b1  = (const float*)d_in[6];
  const float* decWih0 = (const float*)d_in[7];
  const float* decWhh0 = (const float*)d_in[8];
  const float* dec_b0  = (const float*)d_in[9];
  const float* decWih1 = (const float*)d_in[10];
  const float* decWhh1 = (const float*)d_in[11];
  const float* dec_b1  = (const float*)d_in[12];
  const float* lin_w   = (const float*)d_in[13];
  const float* lin_b   = (const float*)d_in[14];

  _Float16* encW0 = (_Float16*)d_ws;                    // G4*K0 halves
  _Float16* encW1 = encW0 + (size_t)G4 * K0;            // G4*K1 halves
  _Float16* decW0 = encW1 + (size_t)G4 * K1;            // G4*K0 halves
  _Float16* decW1 = decW0 + (size_t)G4 * K0;            // G4*K1 halves

  cvt_w0cat<<<(G4 * K0 + 255) / 256, 256, 0, stream>>>(encWhh0, encWih0, encW0);
  cvt_w1cat<<<(G4 * K1 + 255) / 256, 256, 0, stream>>>(encWih1, encWhh1, encW1);
  cvt_w0cat<<<(G4 * K0 + 255) / 256, 256, 0, stream>>>(decWhh0, decWih0, decW0);
  cvt_w1cat<<<(G4 * K1 + 255) / 256, 256, 0, stream>>>(decWih1, decWhh1, decW1);

  lstm_encdec_kernel<<<NBLK, NTHREADS, 0, stream>>>(
      x, enc_b0, enc_b1, dec_b0, dec_b1,
      encW0, encW1, decW0, decW1, lin_w, lin_b, (float*)d_out);
}
